// Attention_7499012899301
// MI455X (gfx1250) — matrix operand order repaired and checked by reading, not compiled, not run on hardware
//
#include <hip/hip_runtime.h>

// ---------- types ----------
typedef __attribute__((ext_vector_type(16))) __bf16        bf16x16;
typedef __attribute__((ext_vector_type(8)))  float         f32x8;
typedef __attribute__((ext_vector_type(4)))  unsigned int  u32x4;

union AFrag { u32x4 q[2]; bf16x16 v; };

__device__ __forceinline__ f32x8 zero8() {
  f32x8 z = {0.f, 0.f, 0.f, 0.f, 0.f, 0.f, 0.f, 0.f};
  return z;
}

// f32 -> bf16 (round to nearest even)
__device__ __forceinline__ unsigned short f2bf(float f) {
  unsigned int u = __float_as_uint(f);
  unsigned int r = u + 0x7FFFu + ((u >> 16) & 1u);
  return (unsigned short)(r >> 16);
}

// ---------- WMMA fragment loaders (wave32, 16x16x32 bf16) ----------
// A matrix 16x32 bf16, row-major source with row stride ld (elements).
// Lane L (0-15): row L, K = 0..7 and 16..23 ; lane L+16: row L, K = 8..15 and 24..31.
__device__ __forceinline__ AFrag load_a(const unsigned short* __restrict__ p, int ld, int lane) {
  AFrag f;
  int row = lane & 15;
  int kb  = (lane >> 4) << 3;                // 0 or 8
  const unsigned short* base = p + (size_t)row * ld + kb;
  f.q[0] = *(const u32x4*)(base);            // K = kb .. kb+7
  f.q[1] = *(const u32x4*)(base + 16);       // K = kb+16 .. kb+23
  return f;
}

// B matrix 32x16 bf16 loaded from "column-contiguous" storage: each output
// column n is a contiguous run of K values with stride ld between columns.
// Lane L (0-15): col L, K = 0..7 and 16..23 ; lane L+16: col L, K = 8..15 and 24..31.
__device__ __forceinline__ AFrag load_b(const unsigned short* __restrict__ p, int ld, int lane) {
  AFrag f;
  int n  = lane & 15;
  int kb = (lane >> 4) << 3;                 // 0 or 8
  const unsigned short* base = p + (size_t)n * ld + kb;
  f.q[0] = *(const u32x4*)(base);
  f.q[1] = *(const u32x4*)(base + 16);
  return f;
}

__device__ __forceinline__ f32x8 wmma_bf16(const AFrag& a, const AFrag& b, f32x8 c) {
  return __builtin_amdgcn_wmma_f32_16x16x32_bf16(false, a.v, false, b.v, (short)0, c, false, false);
}

// ---------- problem constants ----------
#define BATCH 2
#define SEQ   2048
#define EMB   1024
#define NH    16
#define HD    64
#define M_TOT (BATCH * SEQ)                  // 4096 rows
#define JQKV  (3 * NH * HD)                  // 3072 qkv output channels

// ---------- pack kernels ----------
__global__ void k_f2bf(const float* __restrict__ in, unsigned short* __restrict__ out, int ntot) {
  int i = blockIdx.x * blockDim.x + threadIdx.x;
  if (i < ntot) out[i] = f2bf(in[i]);
}

// W_{Q,K,V}[N][E][H] f32 -> Wt[j = which*1024 + n*64 + h][e] bf16
__global__ void k_pack_wqkv(const float* __restrict__ wq, const float* __restrict__ wk,
                            const float* __restrict__ wv, unsigned short* __restrict__ wt) {
  int o = blockIdx.x * blockDim.x + threadIdx.x;     // [0, 3*1024*1024)
  int j = o >> 10;
  int e = o & 1023;
  int which = j >> 10;
  int n = (j >> 6) & 15;
  int h = j & 63;
  const float* w = (which == 0) ? wq : (which == 1) ? wk : wv;
  wt[o] = f2bf(w[((size_t)n * EMB + e) * HD + h]);
}

// W_O[N][H][E] f32 -> WoT[e][j = n*64 + h] bf16
__global__ void k_pack_wo(const float* __restrict__ wo, unsigned short* __restrict__ wot) {
  int o = blockIdx.x * blockDim.x + threadIdx.x;     // [0, 1024*1024) ; o = e*1024 + j
  int e = o >> 10;
  int j = o & 1023;
  wot[o] = f2bf(wo[(size_t)j * EMB + e]);
}

// ---------- QKV projection GEMM ----------
// C[m][j] = sum_e xbf[m][e] * wt[j][e], j in [0,3072). One wave = 16x64 tile.
// Software-pipelined: fragments for step k+1 are in flight during step k WMMAs.
// Writes Q,K as [B][N][T][H] bf16, V transposed as [B][N][H][T] bf16.
__global__ __launch_bounds__(256) void k_qkv(const unsigned short* __restrict__ xbf,
                                             const unsigned short* __restrict__ wt,
                                             const float* __restrict__ bQ,
                                             const float* __restrict__ bK,
                                             const float* __restrict__ bV,
                                             unsigned short* __restrict__ Q,
                                             unsigned short* __restrict__ K,
                                             unsigned short* __restrict__ Vt) {
  const int lane = threadIdx.x & 31;
  const int wave = threadIdx.x >> 5;
  const int wg = blockIdx.x * 8 + wave;              // 12288 wave tiles
  const int mtile = wg / (JQKV / 64);                // 48 j-groups
  const int jg = wg % (JQKV / 64);
  const int which = jg >> 4;                         // 0=Q 1=K 2=V
  const int n = jg & 15;

  f32x8 acc[4];
#pragma unroll
  for (int s = 0; s < 4; ++s) acc[s] = zero8();

  const unsigned short* arow = xbf + (size_t)mtile * 16 * EMB;
  const unsigned short* brow = wt + (size_t)jg * 64 * EMB;

  // prologue: fragments for k0 = 0
  AFrag a_cur = load_a(arow, EMB, lane);
  AFrag b_cur[4];
#pragma unroll
  for (int s = 0; s < 4; ++s) b_cur[s] = load_b(brow + (size_t)s * 16 * EMB, EMB, lane);

  for (int k0 = 0; k0 < EMB - 32; k0 += 32) {
    const int kn = k0 + 32;
    AFrag a_nxt = load_a(arow + kn, EMB, lane);
    AFrag b_nxt[4];
#pragma unroll
    for (int s = 0; s < 4; ++s) b_nxt[s] = load_b(brow + (size_t)s * 16 * EMB + kn, EMB, lane);
#pragma unroll
    for (int s = 0; s < 4; ++s) acc[s] = wmma_bf16(a_cur, b_cur[s], acc[s]);
    a_cur = a_nxt;
#pragma unroll
    for (int s = 0; s < 4; ++s) b_cur[s] = b_nxt[s];
  }
#pragma unroll
  for (int s = 0; s < 4; ++s) acc[s] = wmma_bf16(a_cur, b_cur[s], acc[s]);

  const float* bias = (which == 0) ? bQ : (which == 1) ? bK : bV;
#pragma unroll
  for (int s = 0; s < 4; ++s) {
    int h = s * 16 + (lane & 15);
    float bv = bias[n * HD + h];
#pragma unroll
    for (int r = 0; r < 8; ++r) {
      int m = mtile * 16 + ((lane < 16) ? r : r + 8);
      int b = m >> 11;                               // / SEQ
      int t = m & (SEQ - 1);
      unsigned short o = f2bf(acc[s][r] + bv);
      size_t bn = (size_t)(b * NH + n);
      if (which == 0)      Q[(bn * SEQ + t) * HD + h] = o;
      else if (which == 1) K[(bn * SEQ + t) * HD + h] = o;
      else                 Vt[(bn * HD + h) * SEQ + t] = o;
    }
  }
}

// ---------- flash attention (causal, online softmax) ----------
// One wave = 16 query rows of one (b, n). 8 waves per block.
// V fragments are loaded before the softmax so their latency hides under
// the S WMMAs + softmax VALU/shuffle work + LDS round-trip of P.
__global__ __launch_bounds__(256) void k_attn(const unsigned short* __restrict__ Q,
                                              const unsigned short* __restrict__ K,
                                              const unsigned short* __restrict__ Vt,
                                              unsigned short* __restrict__ Z) {
  __shared__ alignas(16) unsigned short pbuf[8][16 * 32];
  const int lane = threadIdx.x & 31;
  const int wave = threadIdx.x >> 5;

  int bid = blockIdx.x;                              // [0, 2*16*16)
  int qc = bid & 15;
  int n  = (bid >> 4) & 15;
  int b  = bid >> 8;
  int q0 = qc * 128 + wave * 16;

  size_t bn = (size_t)(b * NH + n);
  const unsigned short* Qp = Q + bn * SEQ * HD;
  const unsigned short* Kp = K + bn * SEQ * HD;
  const unsigned short* Vp = Vt + bn * HD * SEQ;
  unsigned short* pl = &pbuf[wave][0];

  // Q fragments for this q-tile (H = 64 -> two K=32 chunks), loaded once.
  AFrag qa0 = load_a(Qp + (size_t)q0 * HD + 0, HD, lane);
  AFrag qa1 = load_a(Qp + (size_t)q0 * HD + 32, HD, lane);

  f32x8 zacc[4];
#pragma unroll
  for (int s = 0; s < 4; ++s) zacc[s] = zero8();
  float mrun[8], lrun[8];
#pragma unroll
  for (int r = 0; r < 8; ++r) { mrun[r] = -__builtin_inff(); lrun[r] = 0.f; }

  const float scale = 0.125f;                        // 1/sqrt(64)
  const int col = lane & 15;
  const int kmax = q0 + 16;                          // causal bound (exclusive)

  for (int k0 = 0; k0 < kmax; k0 += 32) {
    // K fragments for the two S tiles (keys [k0,k0+16) and [k0+16,k0+32))
    AFrag kb0 = load_b(Kp + (size_t)k0 * HD + 0, HD, lane);
    AFrag kb1 = load_b(Kp + (size_t)k0 * HD + 32, HD, lane);
    AFrag kb2 = load_b(Kp + (size_t)(k0 + 16) * HD + 0, HD, lane);
    AFrag kb3 = load_b(Kp + (size_t)(k0 + 16) * HD + 32, HD, lane);
    // V fragments issued early; consumed only after softmax + LDS round-trip.
    AFrag vb[4];
#pragma unroll
    for (int s = 0; s < 4; ++s) vb[s] = load_b(Vp + (size_t)(s * 16) * SEQ + k0, SEQ, lane);

    f32x8 s0 = zero8(), s1 = zero8();
    s0 = wmma_bf16(qa0, kb0, s0);
    s0 = wmma_bf16(qa1, kb1, s0);
    s1 = wmma_bf16(qa0, kb2, s1);
    s1 = wmma_bf16(qa1, kb3, s1);

    // online softmax per row; row r maps to q0 + r (lanes 0-15) / q0 + r + 8 (lanes 16-31)
#pragma unroll
    for (int r = 0; r < 8; ++r) {
      int qrow = q0 + ((lane < 16) ? r : r + 8);
      float v0 = s0[r] * scale;
      float v1 = s1[r] * scale;
      if (k0 + col > qrow)      v0 = -__builtin_inff();
      if (k0 + 16 + col > qrow) v1 = -__builtin_inff();

      float rmax = fmaxf(v0, v1);
#pragma unroll
      for (int msk = 1; msk <= 8; msk <<= 1) rmax = fmaxf(rmax, __shfl_xor(rmax, msk, 32));
      float mnew = fmaxf(mrun[r], rmax);
      float alpha = __expf(mrun[r] - mnew);
      float p0 = __expf(v0 - mnew);
      float p1 = __expf(v1 - mnew);
      float rsum = p0 + p1;
#pragma unroll
      for (int msk = 1; msk <= 8; msk <<= 1) rsum += __shfl_xor(rsum, msk, 32);
      lrun[r] = lrun[r] * alpha + rsum;
      mrun[r] = mnew;
#pragma unroll
      for (int s = 0; s < 4; ++s) zacc[s][r] *= alpha;

      int prow = (lane < 16) ? r : r + 8;
      pl[prow * 32 + col]      = f2bf(p0);
      pl[prow * 32 + col + 16] = f2bf(p1);
    }
    asm volatile("" ::: "memory");                   // order LDS stores vs. re-load

    // P (16x32 bf16) as A-fragment from LDS; PV into z accumulators.
    AFrag pa = load_a(pl, 32, lane);
#pragma unroll
    for (int s = 0; s < 4; ++s) zacc[s] = wmma_bf16(pa, vb[s], zacc[s]);
    asm volatile("" ::: "memory");                   // WAR: next iter rewrites pl
  }

  // normalize + store Z[b][t][n][h] bf16
#pragma unroll
  for (int s = 0; s < 4; ++s) {
    int h = s * 16 + (lane & 15);
#pragma unroll
    for (int r = 0; r < 8; ++r) {
      int t = q0 + ((lane < 16) ? r : r + 8);
      float val = zacc[s][r] / lrun[r];
      Z[((size_t)(b * SEQ + t) * NH + n) * HD + h] = f2bf(val);
    }
  }
}

// ---------- output projection GEMM ----------
// out[m][e] = sum_j zbf[m][j] * wot[e][j] + bO[e]  (software-pipelined)
__global__ __launch_bounds__(256) void k_oproj(const unsigned short* __restrict__ zbf,
                                               const unsigned short* __restrict__ wot,
                                               const float* __restrict__ bO,
                                               float* __restrict__ out) {
  const int lane = threadIdx.x & 31;
  const int wave = threadIdx.x >> 5;
  const int wg = blockIdx.x * 8 + wave;              // 4096 wave tiles
  const int mtile = wg >> 4;
  const int jg = wg & 15;

  f32x8 acc[4];
#pragma unroll
  for (int s = 0; s < 4; ++s) acc[s] = zero8();

  const unsigned short* arow = zbf + (size_t)mtile * 16 * (NH * HD);
  const unsigned short* brow = wot + (size_t)jg * 64 * (NH * HD);

  AFrag a_cur = load_a(arow, NH * HD, lane);
  AFrag b_cur[4];
#pragma unroll
  for (int s = 0; s < 4; ++s) b_cur[s] = load_b(brow + (size_t)s * 16 * (NH * HD), NH * HD, lane);

  for (int k0 = 0; k0 < NH * HD - 32; k0 += 32) {
    const int kn = k0 + 32;
    AFrag a_nxt = load_a(arow + kn, NH * HD, lane);
    AFrag b_nxt[4];
#pragma unroll
    for (int s = 0; s < 4; ++s) b_nxt[s] = load_b(brow + (size_t)s * 16 * (NH * HD) + kn, NH * HD, lane);
#pragma unroll
    for (int s = 0; s < 4; ++s) acc[s] = wmma_bf16(a_cur, b_cur[s], acc[s]);
    a_cur = a_nxt;
#pragma unroll
    for (int s = 0; s < 4; ++s) b_cur[s] = b_nxt[s];
  }
#pragma unroll
  for (int s = 0; s < 4; ++s) acc[s] = wmma_bf16(a_cur, b_cur[s], acc[s]);

#pragma unroll
  for (int s = 0; s < 4; ++s) {
    int e = jg * 64 + s * 16 + (lane & 15);
    float be = bO[e];
#pragma unroll
    for (int r = 0; r < 8; ++r) {
      int m = mtile * 16 + ((lane < 16) ? r : r + 8);
      out[(size_t)m * EMB + e] = acc[s][r] + be;
    }
  }
}

// ---------- host launch ----------
extern "C" void kernel_launch(void* const* d_in, const int* in_sizes, int n_in,
                              void* d_out, int out_size, void* d_ws, size_t ws_size,
                              hipStream_t stream) {
  const float* x  = (const float*)d_in[0];
  const float* wq = (const float*)d_in[1];
  const float* wk = (const float*)d_in[2];
  const float* wv = (const float*)d_in[3];
  const float* wo = (const float*)d_in[4];
  const float* bq = (const float*)d_in[5];
  const float* bk = (const float*)d_in[6];
  const float* bv = (const float*)d_in[7];
  const float* bo = (const float*)d_in[8];
  float* out = (float*)d_out;

  char* ws = (char*)d_ws;
  unsigned short* xbf = (unsigned short*)ws; ws += (size_t)M_TOT * EMB * 2;        // 8 MiB
  unsigned short* wt  = (unsigned short*)ws; ws += (size_t)JQKV * EMB * 2;         // 6 MiB
  unsigned short* wot = (unsigned short*)ws; ws += (size_t)EMB * NH * HD * 2;      // 2 MiB
  unsigned short* Qb  = (unsigned short*)ws; ws += (size_t)BATCH * NH * SEQ * HD * 2; // 8 MiB
  unsigned short* Kb  = (unsigned short*)ws; ws += (size_t)BATCH * NH * SEQ * HD * 2; // 8 MiB
  unsigned short* Vtb = (unsigned short*)ws; ws += (size_t)BATCH * NH * HD * SEQ * 2; // 8 MiB
  unsigned short* Zb  = (unsigned short*)ws;                                          // 8 MiB

  k_f2bf<<<(M_TOT * EMB) / 256, 256, 0, stream>>>(x, xbf, M_TOT * EMB);
  k_pack_wqkv<<<(JQKV * EMB) / 256, 256, 0, stream>>>(wq, wk, wv, wt);
  k_pack_wo<<<(EMB * NH * HD) / 256, 256, 0, stream>>>(wo, wot);

  // 4096/16 m-tiles * 48 j-groups = 12288 waves, 8 waves/block
  k_qkv<<<1536, 256, 0, stream>>>(xbf, wt, bq, bk, bv, Qb, Kb, Vtb);

  // B * N * (T/128) blocks, 8 waves each covering 16 q-rows
  k_attn<<<BATCH * NH * (SEQ / 128), 256, 0, stream>>>(Qb, Kb, Vtb, Zb);

  // 256 m-tiles * 16 j-groups = 4096 waves, 8 waves/block
  k_oproj<<<512, 256, 0, stream>>>(Zb, wot, bo, out);
}
